// GetContext_11287174053943
// MI455X (gfx1250) — matrix operand order repaired and checked by reading, not compiled, not run on hardware
//
#include <hip/hip_runtime.h>
#include <hip/hip_bf16.h>
#include <stdint.h>

// ---------------------------------------------------------------------------
// AttentiveFP GetContext for MI455X (gfx1250, wave32, WMMA bf16)
// N=25000 nodes, E=500000 edges, ND=32, ED=19, G=200 (padded to GP=224)
// ---------------------------------------------------------------------------

#define NNODES 25000
#define NEDGES 500000
#define ND_ 32
#define ED_ 19
#define G_  200
#define GP_ 224            // G padded to multiple of 32 (7 k-steps, 14 n-tiles)
#define KE_ 64             // ND+ED=51 padded to 64 (2 k-steps)
#define NT_FULL 14         // GP/16
#define NT_G    13         // ceil(G/16): n-tiles that contain real columns

typedef __bf16 bf16_t;
typedef __attribute__((ext_vector_type(16))) __bf16 v16bf;
typedef __attribute__((ext_vector_type(8)))  __bf16 v8bf;
typedef __attribute__((ext_vector_type(8)))  float  v8f;

__device__ __forceinline__ v8f wmma_bf16_acc(v16bf a, v16bf b, v8f c) {
  // D = A(16x32 bf16) * B(32x16 bf16) + C(16x16 f32)
  return __builtin_amdgcn_wmma_f32_16x16x32_bf16(
      /*neg_a=*/false, a, /*neg_b=*/false, b,
      /*c_mod=*/(short)0, c, /*reuse_a=*/false, /*reuse_b=*/false);
}

// A-fragment (16x32 bf16, MxK): lane half h holds K = h*16+[0..7] in elems 0..7
// and K = h*16+8+[0..7] in elems 8..15 (contiguous, matches B loads).
union AFrag { v16bf v; v8bf h[2]; };

__device__ __forceinline__ v16bf load_a_bf(const bf16_t* rowk, int half) {
  AFrag u;
  u.h[0] = *(const v8bf*)(rowk + half * 16);
  u.h[1] = *(const v8bf*)(rowk + half * 16 + 8);
  return u.v;
}

// ------------------------- weight packing ----------------------------------
__global__ void k_pack_simple(const float* __restrict__ w, bf16_t* __restrict__ dst,
                              int Ns, int Ks, int Np, int Kp) {
  int idx = blockIdx.x * blockDim.x + threadIdx.x;
  if (idx >= Np * Kp) return;
  int n = idx / Kp, k = idx % Kp;
  float v = (n < Ns && k < Ks) ? w[n * Ks + k] : 0.0f;
  dst[idx] = (bf16_t)v;
}

// W_ih / W_hh [600,200] -> [3*GP, GP] with gate g occupying rows g*GP..
__global__ void k_pack_gates(const float* __restrict__ w, bf16_t* __restrict__ dst) {
  int idx = blockIdx.x * blockDim.x + threadIdx.x;
  if (idx >= 3 * GP_ * GP_) return;
  int row = idx / GP_, k = idx % GP_;
  int g = row / GP_, r = row % GP_;
  float v = (r < G_ && k < G_) ? w[(g * G_ + r) * G_ + k] : 0.0f;
  dst[idx] = (bf16_t)v;
}

__global__ void k_init(unsigned* __restrict__ nmax, float* __restrict__ ndenom,
                       float* __restrict__ ctx) {
  size_t i = (size_t)blockIdx.x * blockDim.x + threadIdx.x;
  if (i < NNODES) { nmax[i] = 0u; ndenom[i] = 0.0f; }
  if (i < (size_t)NNODES * GP_) ctx[i] = 0.0f;
}

// ------------------- hv_new = lrelu(node_feats @ Wn^T + bn) ----------------
__global__ void k_node_proj(const float* __restrict__ nf, const bf16_t* __restrict__ Wn_b,
                            const float* __restrict__ bn,
                            float* __restrict__ hv_f, bf16_t* __restrict__ hv_bf) {
  const int MT = (NNODES + 15) / 16;
  int wid = (blockIdx.x * blockDim.x + threadIdx.x) >> 5;
  int lane = threadIdx.x & 31;
  if (wid >= MT * NT_FULL) return;               // wave-uniform
  int mt = wid / NT_FULL, nt = wid % NT_FULL;
  int mBase = mt * 16, nBase = nt * 16;
  int half = lane >> 4, l16 = lane & 15;

  int arow = mBase + l16; if (arow >= NNODES) arow = NNODES - 1;
  const float* ap = nf + (size_t)arow * ND_;     // K = 32 exactly, one step
  v16bf a;
#pragma unroll
  for (int j = 0; j < 8; ++j) {
    a[j]     = (bf16_t)ap[half * 16 + j];
    a[8 + j] = (bf16_t)ap[half * 16 + 8 + j];
  }
  v16bf b = *(const v16bf*)(Wn_b + (size_t)(nBase + l16) * ND_ + half * 16);
  v8f c = {};
  c = wmma_bf16_acc(a, b, c);

  int cc = nBase + l16;
  float bias = (cc < G_) ? bn[cc] : 0.0f;        // hoisted, one load
  if (mBase + 16 <= NNODES) {                    // full tile: straight-line stores
#pragma unroll
    for (int r = 0; r < 8; ++r) {
      int m = mBase + r + half * 8;
      float acc = c[r] + bias;
      float y = acc > 0.0f ? acc : 0.01f * acc;
      hv_f[(size_t)m * GP_ + cc]  = y;
      hv_bf[(size_t)m * GP_ + cc] = (bf16_t)y;
    }
  } else {
#pragma unroll
    for (int r = 0; r < 8; ++r) {
      int m = mBase + r + half * 8;
      if (m < NNODES) {
        float acc = c[r] + bias;
        float y = acc > 0.0f ? acc : 0.01f * acc;
        hv_f[(size_t)m * GP_ + cc]  = y;
        hv_bf[(size_t)m * GP_ + cc] = (bf16_t)y;
      }
    }
  }
}

// ---- he1 = lrelu(cat(node_feats[src], edge_feats) @ We1^T + be1), bf16 ----
__global__ void k_edge_proj(const float* __restrict__ nf, const float* __restrict__ ef,
                            const int* __restrict__ src, const bf16_t* __restrict__ We1_b,
                            const float* __restrict__ be1, bf16_t* __restrict__ he1) {
  const int MT = NEDGES / 16;                    // 31250 exact
  int wid = (blockIdx.x * blockDim.x + threadIdx.x) >> 5;
  int lane = threadIdx.x & 31;
  if (wid >= MT * NT_FULL) return;
  int mt = wid / NT_FULL, nt = wid % NT_FULL;
  int mBase = mt * 16, nBase = nt * 16;
  int half = lane >> 4, l16 = lane & 15;

  int erow = mBase + l16;                        // always < NEDGES
  int s = src[erow];
  const float* np_ = nf + (size_t)s * ND_;
  const float* ep_ = ef + (size_t)erow * ED_;

  // k-step 0: source-node features (K 0..31)
  v16bf a0;
#pragma unroll
  for (int j = 0; j < 8; ++j) {
    a0[j]     = (bf16_t)np_[half * 16 + j];
    a0[8 + j] = (bf16_t)np_[half * 16 + 8 + j];
  }
  // k-step 1: edge features (K 32..63, cols >= 19 are zero pad)
  v16bf a1;
#pragma unroll
  for (int j = 0; j < 8; ++j) {
    int k0 = half * 16 + j;
    int k1 = half * 16 + 8 + j;
    a1[j]     = (k0 < ED_) ? (bf16_t)ep_[k0] : (bf16_t)0.0f;
    a1[8 + j] = (k1 < ED_) ? (bf16_t)ep_[k1] : (bf16_t)0.0f;
  }

  const bf16_t* wrow = We1_b + (size_t)(nBase + l16) * KE_;
  v16bf b0 = *(const v16bf*)(wrow + half * 16);
  v16bf b1 = *(const v16bf*)(wrow + 32 + half * 16);

  v8f c = {};
  c = wmma_bf16_acc(a0, b0, c);
  c = wmma_bf16_acc(a1, b1, c);

  int cc = nBase + l16;
  float bias = (cc < G_) ? be1[cc] : 0.0f;
#pragma unroll
  for (int r = 0; r < 8; ++r) {
    int m = mBase + r + half * 8;
    float acc = c[r] + bias;
    float y = acc > 0.0f ? acc : 0.01f * acc;    // padded cols -> lrelu(0)=0
    he1[(size_t)m * GP_ + cc] = (bf16_t)y;
  }
}

// ---- logits[e] = lrelu([hv_new[dst], he1] . We2 + be2); seg-max via atomics
__global__ void k_logits(const float* __restrict__ hv_f, const bf16_t* __restrict__ he1,
                         const int* __restrict__ dst, const float* __restrict__ We2,
                         const float* __restrict__ be2, float* __restrict__ exl,
                         unsigned* __restrict__ nmax) {
  int wid = (blockIdx.x * blockDim.x + threadIdx.x) >> 5;   // one wave per edge
  int lane = threadIdx.x & 31;
  if (wid >= NEDGES) return;
  int d = dst[wid];
  const float*  hp = hv_f + (size_t)d * GP_;
  const bf16_t* ep = he1 + (size_t)wid * GP_;
  float acc = 0.0f;
  for (int c = lane; c < G_; c += 32)
    acc += We2[c] * hp[c] + We2[G_ + c] * (float)ep[c];
#pragma unroll
  for (int s = 16; s >= 1; s >>= 1) acc += __shfl_xor(acc, s, 32);
  if (lane == 0) {
    float lg = acc + be2[0];
    lg = lg > 0.0f ? lg : 0.01f * lg;
    exl[wid] = lg;
    unsigned kb = __float_as_uint(lg);
    kb = (kb & 0x80000000u) ? ~kb : (kb | 0x80000000u);   // order-preserving key
    atomicMax(nmax + d, kb);
  }
}

__global__ void k_expsum(const int* __restrict__ dst, float* __restrict__ exl,
                         const unsigned* __restrict__ nmax, float* __restrict__ ndenom) {
  int e = blockIdx.x * blockDim.x + threadIdx.x;
  if (e >= NEDGES) return;
  int d = dst[e];
  unsigned key = nmax[d];
  unsigned b = (key & 0x80000000u) ? (key & 0x7FFFFFFFu) : ~key;
  float m = __uint_as_float(b);
  float ex = __expf(exl[e] - m);
  exl[e] = ex;
  atomicAdd(ndenom + d, ex);
}

// ---- big GEMM: e = softmax * (he1 @ Wet^T + bet); scatter-add into ctx ----
// One block (8 waves) per 16-edge m-tile. The A-tile (16 rows x 448B, fully
// contiguous in he1) is staged into LDS with GLOBAL_LOAD_ASYNC_TO_LDS_B128
// (ASYNCcnt path), then all 13 n-tiles feed WMMA from ds_load.
__global__ void __launch_bounds__(256)
k_edge_attn(const bf16_t* __restrict__ he1, const bf16_t* __restrict__ Wet_b,
            const float* __restrict__ bet, const int* __restrict__ dst,
            const float* __restrict__ exl, const float* __restrict__ ndenom,
            float* __restrict__ ctx) {
  __shared__ __align__(32) bf16_t aTile[16 * GP_];   // 7168 B
  __shared__ float aCoef[16];
  __shared__ int   dRow[16];

  int mBase = blockIdx.x * 16;                 // grid = NEDGES/16 blocks
  int t = threadIdx.x;
  int wave = t >> 5, lane = t & 31;
  int half = lane >> 4, l16 = lane & 15;

  // ---- async stage: 448 chunks of 16B, contiguous global region ----
  {
    const bf16_t* gbase = he1 + (size_t)mBase * GP_;
    unsigned ldsBase = (unsigned)(uintptr_t)(&aTile[0]);
    unsigned lo0 = ldsBase + (unsigned)t * 16u;
    const bf16_t* gp0 = gbase + t * 8;
    asm volatile("global_load_async_to_lds_b128 %0, %1, off"
                 :: "v"(lo0), "v"(gp0) : "memory");
    int ch2 = t + 256;
    if (ch2 < 448) {
      unsigned lo1 = ldsBase + (unsigned)ch2 * 16u;
      const bf16_t* gp1 = gbase + ch2 * 8;
      asm volatile("global_load_async_to_lds_b128 %0, %1, off"
                   :: "v"(lo1), "v"(gp1) : "memory");
    }
    asm volatile("s_wait_asynccnt 0" ::: "memory");
  }
  // per-row softmax coefficient + destination, computed once per block
  if (t < 16) {
    int m = mBase + t;
    int d = dst[m];
    dRow[t]  = d;
    aCoef[t] = exl[m] / ndenom[d];
  }
  __syncthreads();

  for (int nt = wave; nt < NT_G; nt += 8) {    // wave-uniform loop
    int nBase = nt * 16;
    const bf16_t* brow = Wet_b + (size_t)(nBase + l16) * GP_;
    const bf16_t* asrc = &aTile[l16 * GP_];
    v8f c = {};
#pragma unroll
    for (int kt = 0; kt < GP_ / 32; ++kt) {    // 7 k-steps
      int kb = kt * 32;
      v16bf a = load_a_bf(asrc + kb, half);    // ds_load from LDS tile
      v16bf b = *(const v16bf*)(brow + kb + half * 16);
      c = wmma_bf16_acc(a, b, c);
    }
    int cc = nBase + l16;
    if (cc < G_) {
      float bv = bet[cc];
#pragma unroll
      for (int r = 0; r < 8; ++r) {
        int rr = r + half * 8;
        atomicAdd(ctx + (size_t)dRow[rr] * GP_ + cc, aCoef[rr] * (c[r] + bv));
      }
    }
  }
}

__global__ void k_elu(const float* __restrict__ ctx, bf16_t* __restrict__ ctx_bf) {
  size_t i = (size_t)blockIdx.x * blockDim.x + threadIdx.x;
  if (i >= (size_t)NNODES * GP_) return;
  float v = ctx[i];
  float e = v > 0.0f ? v : (__expf(v) - 1.0f);   // elu(0)=0 keeps padding clean
  ctx_bf[i] = (bf16_t)e;
}

// ---- fused GRU: 6 WMMA accumulators per tile, writes relu(h) to d_out ----
__global__ void k_gru(const bf16_t* __restrict__ ctx_bf, const bf16_t* __restrict__ hv_bf,
                      const float* __restrict__ hv_f,
                      const bf16_t* __restrict__ Wih_b, const bf16_t* __restrict__ Whh_b,
                      const float* __restrict__ b_ih, const float* __restrict__ b_hh,
                      float* __restrict__ out) {
  const int MT = (NNODES + 15) / 16;
  int wid = (blockIdx.x * blockDim.x + threadIdx.x) >> 5;
  int lane = threadIdx.x & 31;
  if (wid >= MT * NT_G) return;
  int mt = wid / NT_G, nt = wid % NT_G;
  int mBase = mt * 16, nBase = nt * 16;
  int half = lane >> 4, l16 = lane & 15;

  int arow = mBase + l16; if (arow >= NNODES) arow = NNODES - 1;
  const bf16_t* ap_c = ctx_bf + (size_t)arow * GP_;
  const bf16_t* ap_h = hv_bf + (size_t)arow * GP_;
  __builtin_prefetch(ap_c, 0, 1);              // global_prefetch_b8
  __builtin_prefetch(ap_h, 0, 1);
  size_t wrow = (size_t)(nBase + l16) * GP_;
  const size_t GATE = (size_t)GP_ * GP_;

  v8f cir = {}, ciz = {}, cin = {}, chr_ = {}, chz = {}, chn = {};
#pragma unroll
  for (int kt = 0; kt < GP_ / 32; ++kt) {
    int kb = kt * 32;
    v16bf ac = load_a_bf(ap_c + kb, half);
    v16bf ah = load_a_bf(ap_h + kb, half);
    v16bf b;
    b = *(const v16bf*)(Wih_b + wrow + kb + half * 16);            cir  = wmma_bf16_acc(ac, b, cir);
    b = *(const v16bf*)(Wih_b + GATE + wrow + kb + half * 16);     ciz  = wmma_bf16_acc(ac, b, ciz);
    b = *(const v16bf*)(Wih_b + 2 * GATE + wrow + kb + half * 16); cin  = wmma_bf16_acc(ac, b, cin);
    b = *(const v16bf*)(Whh_b + wrow + kb + half * 16);            chr_ = wmma_bf16_acc(ah, b, chr_);
    b = *(const v16bf*)(Whh_b + GATE + wrow + kb + half * 16);     chz  = wmma_bf16_acc(ah, b, chz);
    b = *(const v16bf*)(Whh_b + 2 * GATE + wrow + kb + half * 16); chn  = wmma_bf16_acc(ah, b, chn);
  }

  int cc = nBase + l16;
  if (cc < G_) {
    float bir = b_ih[cc], biz = b_ih[G_ + cc], bin_ = b_ih[2 * G_ + cc];
    float bhr = b_hh[cc], bhz = b_hh[G_ + cc], bhn = b_hh[2 * G_ + cc];
    bool full = (mBase + 16 <= NNODES);
#pragma unroll
    for (int r = 0; r < 8; ++r) {
      int m = mBase + r + half * 8;
      if (full || m < NNODES) {
        float rr = 1.0f / (1.0f + __expf(-((cir[r] + bir) + (chr_[r] + bhr))));
        float zz = 1.0f / (1.0f + __expf(-((ciz[r] + biz) + (chz[r] + bhz))));
        float nn = tanhf((cin[r] + bin_) + rr * (chn[r] + bhn));
        float hv = hv_f[(size_t)m * GP_ + cc];
        float h = (1.0f - zz) * nn + zz * hv;
        out[(size_t)m * G_ + cc] = h > 0.0f ? h : 0.0f;
      }
    }
  }
}

// ---------------------------------------------------------------------------
extern "C" void kernel_launch(void* const* d_in, const int* in_sizes, int n_in,
                              void* d_out, int out_size, void* d_ws, size_t ws_size,
                              hipStream_t stream) {
  const float* node_feats = (const float*)d_in[0];
  const float* edge_feats = (const float*)d_in[1];
  const int*   srcI       = (const int*)d_in[2];
  const int*   dstI       = (const int*)d_in[3];
  const float* Wn   = (const float*)d_in[4];
  const float* bn   = (const float*)d_in[5];
  const float* We1  = (const float*)d_in[6];
  const float* be1  = (const float*)d_in[7];
  const float* We2  = (const float*)d_in[8];
  const float* be2  = (const float*)d_in[9];
  const float* Wet  = (const float*)d_in[10];
  const float* bet  = (const float*)d_in[11];
  const float* W_ih = (const float*)d_in[12];
  const float* b_ih = (const float*)d_in[13];
  const float* W_hh = (const float*)d_in[14];
  const float* b_hh = (const float*)d_in[15];
  float* out = (float*)d_out;

  size_t off = 0;
  auto take = [&](size_t bytes) -> void* {
    off = (off + 255) & ~(size_t)255;
    void* p = (char*)d_ws + off;
    off += bytes;
    return p;
  };
  bf16_t*   Wn_b   = (bf16_t*)take(sizeof(bf16_t) * GP_ * ND_);
  bf16_t*   We1_b  = (bf16_t*)take(sizeof(bf16_t) * GP_ * KE_);
  bf16_t*   Wet_b  = (bf16_t*)take(sizeof(bf16_t) * GP_ * GP_);
  bf16_t*   Wih_b  = (bf16_t*)take(sizeof(bf16_t) * 3 * GP_ * GP_);
  bf16_t*   Whh_b  = (bf16_t*)take(sizeof(bf16_t) * 3 * GP_ * GP_);
  float*    hv_f   = (float*)take(sizeof(float) * (size_t)NNODES * GP_);
  bf16_t*   hv_bf  = (bf16_t*)take(sizeof(bf16_t) * (size_t)NNODES * GP_);
  bf16_t*   he1    = (bf16_t*)take(sizeof(bf16_t) * (size_t)NEDGES * GP_);
  float*    exl    = (float*)take(sizeof(float) * NEDGES);
  unsigned* nmax   = (unsigned*)take(sizeof(unsigned) * NNODES);
  float*    ndenom = (float*)take(sizeof(float) * NNODES);
  float*    ctx    = (float*)take(sizeof(float) * (size_t)NNODES * GP_);
  bf16_t*   ctx_bf = (bf16_t*)take(sizeof(bf16_t) * (size_t)NNODES * GP_);

  const int TB = 256;  // 8 wave32s per block

  k_pack_simple<<<(GP_ * ND_ + TB - 1) / TB, TB, 0, stream>>>(Wn, Wn_b, G_, ND_, GP_, ND_);
  k_pack_simple<<<(GP_ * KE_ + TB - 1) / TB, TB, 0, stream>>>(We1, We1_b, G_, ND_ + ED_, GP_, KE_);
  k_pack_simple<<<(GP_ * GP_ + TB - 1) / TB, TB, 0, stream>>>(Wet, Wet_b, G_, G_, GP_, GP_);
  k_pack_gates<<<(3 * GP_ * GP_ + TB - 1) / TB, TB, 0, stream>>>(W_ih, Wih_b);
  k_pack_gates<<<(3 * GP_ * GP_ + TB - 1) / TB, TB, 0, stream>>>(W_hh, Whh_b);

  {
    size_t tot = (size_t)NNODES * GP_;
    k_init<<<(int)((tot + TB - 1) / TB), TB, 0, stream>>>(nmax, ndenom, ctx);
  }
  {
    long tiles = (long)((NNODES + 15) / 16) * NT_FULL;
    k_node_proj<<<(int)((tiles * 32 + TB - 1) / TB), TB, 0, stream>>>(node_feats, Wn_b, bn, hv_f, hv_bf);
  }
  {
    long tiles = (long)(NEDGES / 16) * NT_FULL;
    k_edge_proj<<<(int)((tiles * 32 + TB - 1) / TB), TB, 0, stream>>>(node_feats, edge_feats, srcI, We1_b, be1, he1);
  }
  k_logits<<<(int)(((long)NEDGES * 32 + TB - 1) / TB), TB, 0, stream>>>(hv_f, he1, dstI, We2, be2, exl, nmax);
  k_expsum<<<(NEDGES + TB - 1) / TB, TB, 0, stream>>>(dstI, exl, nmax, ndenom);

  // one block per 16-edge m-tile (async LDS staging inside)
  k_edge_attn<<<NEDGES / 16, TB, 0, stream>>>(he1, Wet_b, bet, dstI, exl, ndenom, ctx);

  k_elu<<<(int)(((size_t)NNODES * GP_ + TB - 1) / TB), TB, 0, stream>>>(ctx, ctx_bf);
  {
    long tiles = (long)((NNODES + 15) / 16) * NT_G;
    k_gru<<<(int)((tiles * 32 + TB - 1) / TB), TB, 0, stream>>>(ctx_bf, hv_bf, hv_f, Wih_b, Whh_b, b_ih, b_hh, out);
  }
  (void)in_sizes; (void)n_in; (void)out_size; (void)ws_size;
}
